// MultiHeadAttention_68607807586767
// MI455X (gfx1250) — matrix operand order repaired and checked by reading, not compiled, not run on hardware
//
#include <hip/hip_runtime.h>

#define S_LEN 4096
#define DMODEL 1024
#define NHEADS 16
#define DK 64

typedef __bf16 bf16;
typedef bf16 bf16x8  __attribute__((ext_vector_type(8)));
typedef bf16 bf16x16 __attribute__((ext_vector_type(16)));
typedef float floatx8 __attribute__((ext_vector_type(8)));

__device__ __forceinline__ bf16 f2bf(float f) {
  unsigned u = __builtin_bit_cast(unsigned, f);
  unsigned r = u + 0x7FFFu + ((u >> 16) & 1u);     // round-to-nearest-even
  unsigned short h = (unsigned short)(r >> 16);
  return __builtin_bit_cast(bf16, h);
}

__device__ __forceinline__ bf16x16 make_frag(bf16x8 lo, bf16x8 hi) {
  bf16x16 f;
#pragma unroll
  for (int i = 0; i < 8; ++i) { f[i] = lo[i]; f[8 + i] = hi[i]; }
  return f;
}

// A-matrix 16x32 fragment from row-major [M][K]; p -> element [tile_row0][k0]
__device__ __forceinline__ bf16x16 load_a_frag(const bf16* p, int ld) {
  int lane = threadIdx.x & 31;
  const bf16* q = p + (lane & 15) * ld + ((lane >> 4) << 3);   // K base 0 or 8
  return make_frag(*(const bf16x8*)q, *(const bf16x8*)(q + 16));
}

// B-matrix 32x16 fragment from N-major storage [N][K]; p -> element [tile_n0][k0]
__device__ __forceinline__ bf16x16 load_b_frag(const bf16* p, int ld) {
  int lane = threadIdx.x & 31;
  const bf16* q = p + (lane & 15) * ld + ((lane >> 4) << 3);   // K base 0 or 8
  return make_frag(*(const bf16x8*)q, *(const bf16x8*)(q + 16));
}

__device__ __forceinline__ floatx8 wmma_bf16(bf16x16 a, bf16x16 b, floatx8 c) {
  return __builtin_amdgcn_wmma_f32_16x16x32_bf16(false, a, false, b, (short)0, c,
                                                 false, false);
}

// xor-shuffle via ds_swizzle (group-of-32: and=0x1f, or=0, xor=MASK)
template <int MASK>
__device__ __forceinline__ float swz_xor(float v) {
  int i = __builtin_bit_cast(int, v);
  i = __builtin_amdgcn_ds_swizzle(i, 0x1f | (MASK << 10));
  return __builtin_bit_cast(float, i);
}

// ---------------- fp32 -> bf16 cast (vectorized, 8 elems/thread) ----------------
__global__ void cast_bf16_kernel(const float* __restrict__ src, bf16* __restrict__ dst,
                                 int n8) {
  int t = blockIdx.x * blockDim.x + threadIdx.x;
  if (t >= n8) return;
  const float4* s = (const float4*)src + (size_t)t * 2;
  float4 a = s[0], b = s[1];
  bf16x8 o;
  o[0] = f2bf(a.x); o[1] = f2bf(a.y); o[2] = f2bf(a.z); o[3] = f2bf(a.w);
  o[4] = f2bf(b.x); o[5] = f2bf(b.y); o[6] = f2bf(b.z); o[7] = f2bf(b.w);
  ((bf16x8*)dst)[t] = o;
}

// ---------------- C-tile store helpers (C layout: vgpr i -> row rbase+i) --------
__device__ __forceinline__ void store_qk_tile(bf16* __restrict__ out,
                                              const float* __restrict__ bias,
                                              floatx8 acc, int row0, int col0,
                                              float oscale) {
  int lane = threadIdx.x & 31;
  int col = col0 + (lane & 15);
  float bv = bias[col];
  int h = col >> 6, d = col & 63;
  int rbase = row0 + ((lane >> 4) << 3);
  bf16* o = out + ((size_t)h * S_LEN) * DK + d;
#pragma unroll
  for (int i = 0; i < 8; ++i)
    o[(size_t)(rbase + i) * DK] = f2bf((acc[i] + bv) * oscale);
}

__device__ __forceinline__ void store_vt_tile(bf16* __restrict__ out,
                                              const float* __restrict__ bias,
                                              floatx8 acc, int row0, int col0) {
  int lane = threadIdx.x & 31;
  int col = col0 + (lane & 15);
  float bv = bias[col];
  int h = col >> 6, d = col & 63;
  int rbase = row0 + ((lane >> 4) << 3);
  bf16x8 o8;
#pragma unroll
  for (int i = 0; i < 8; ++i) o8[i] = f2bf(acc[i] + bv);
  *(bf16x8*)(out + ((size_t)(h * DK + d)) * S_LEN + rbase) = o8;
}

__device__ __forceinline__ void store_f32_tile(float* __restrict__ out,
                                               const float* __restrict__ bias,
                                               floatx8 acc, int row0, int col0) {
  int lane = threadIdx.x & 31;
  int col = col0 + (lane & 15);
  float bv = bias[col];
  int rbase = row0 + ((lane >> 4) << 3);
#pragma unroll
  for (int i = 0; i < 8; ++i)
    out[(size_t)(rbase + i) * DMODEL + col] = acc[i] + bv;
}

// ---------------- projection GEMM: Y = (X @ W^T + b) * oscale, 32x32/wave -------
// mode 0: bf16 head-major out[h][s][d] (Q, K);  mode 1: bf16 transposed out[h][d][s]
__global__ void __launch_bounds__(256)
proj_bf16_kernel(const bf16* __restrict__ X, const bf16* __restrict__ W,
                 const float* __restrict__ bias, bf16* __restrict__ out, int mode,
                 float oscale) {
  int wave = threadIdx.x >> 5;
  int gid = blockIdx.x * 8 + wave;
  int tm = (gid >> 5) << 5;            // row base: 128 tiles of 32
  int tn = (gid & 31) << 5;            // col base: 32 tiles of 32
  const bf16* x0 = X + (size_t)tm * DMODEL;
  const bf16* x1 = x0 + 16 * DMODEL;
  const bf16* w0 = W + (size_t)tn * DMODEL;
  const bf16* w1 = w0 + 16 * DMODEL;

  floatx8 a00 = {}, a01 = {}, a10 = {}, a11 = {};
#pragma unroll 2
  for (int k = 0; k < DMODEL; k += 32) {
    bf16x16 fa0 = load_a_frag(x0 + k, DMODEL);
    bf16x16 fa1 = load_a_frag(x1 + k, DMODEL);
    bf16x16 fb0 = load_b_frag(w0 + k, DMODEL);
    bf16x16 fb1 = load_b_frag(w1 + k, DMODEL);
    a00 = wmma_bf16(fa0, fb0, a00);
    a01 = wmma_bf16(fa0, fb1, a01);
    a10 = wmma_bf16(fa1, fb0, a10);
    a11 = wmma_bf16(fa1, fb1, a11);
  }
  if (mode == 0) {
    store_qk_tile(out, bias, a00, tm, tn, oscale);
    store_qk_tile(out, bias, a01, tm, tn + 16, oscale);
    store_qk_tile(out, bias, a10, tm + 16, tn, oscale);
    store_qk_tile(out, bias, a11, tm + 16, tn + 16, oscale);
  } else {
    store_vt_tile(out, bias, a00, tm, tn);
    store_vt_tile(out, bias, a01, tm, tn + 16);
    store_vt_tile(out, bias, a10, tm + 16, tn);
    store_vt_tile(out, bias, a11, tm + 16, tn + 16);
  }
}

// ---------------- flash attention -------------------------------------------------
// One block = 8 waves = 128 consecutive query rows of one head.
// K/V tiles (32 keys) staged cooperatively in LDS, shared by all 8 waves.
// Q is pre-scaled by (1/sqrt(dk))*log2(e): softmax computed in exp2 domain.
__global__ void __launch_bounds__(256)
flash_attn_kernel(const bf16* __restrict__ Qh, const bf16* __restrict__ Kh,
                  const bf16* __restrict__ Vt, bf16* __restrict__ O) {
  __shared__ bf16 lds_k[32 * DK];         // [key 0..31][d 0..63]     4 KB
  __shared__ bf16 lds_v[DK * 32];         // [d 0..63][key 0..31]     4 KB
  __shared__ bf16 lds_p[8][16 * 32];      // per-wave P staging       8 KB

  int wave = threadIdx.x >> 5;
  int lane = threadIdx.x & 31;
  int tid = threadIdx.x;
  int h = blockIdx.x >> 5;                // 16 heads x 32 blocks/head
  int qt = ((blockIdx.x & 31) << 3) + wave;   // q tile 0..255

  const bf16* qbase = Qh + ((size_t)h * S_LEN + qt * 16) * DK;
  bf16x16 qa0 = load_a_frag(qbase, DK);       // d = 0..31
  bf16x16 qa1 = load_a_frag(qbase + 32, DK);  // d = 32..63

  bf16x16 ones;                               // all-ones B fragment for row sums
#pragma unroll
  for (int i = 0; i < 16; ++i) ones[i] = f2bf(1.0f);

  float m_i[8], l_i[8];
  floatx8 acc[4];
#pragma unroll
  for (int i = 0; i < 8; ++i) { m_i[i] = -3.0e38f; l_i[i] = 0.0f; }
#pragma unroll
  for (int n = 0; n < 4; ++n) acc[n] = floatx8{};

  const bf16* kg_head = Kh + (size_t)h * S_LEN * DK;
  const bf16* vg_head = Vt + (size_t)h * DK * S_LEN;

  for (int jt = 0; jt < S_LEN / 32; ++jt) {
    // ---- cooperative stage of K (32x64) and V^T (64x32) tiles into LDS ----
    {
      const bf16* kg = kg_head + (size_t)(jt * 32) * DK;   // contiguous 4 KB
      *(bf16x8*)(lds_k + tid * 8) = *(const bf16x8*)(kg + tid * 8);
      int vr = tid >> 2, vc = (tid & 3) * 8;               // 64 rows x 4 chunks
      const bf16* vg = vg_head + (size_t)vr * S_LEN + jt * 32 + vc;
      *(bf16x8*)(lds_v + vr * 32 + vc) = *(const bf16x8*)vg;
    }
    __syncthreads();

    // ---- scores (already x log2e/sqrt(dk)): two 16x16 tiles over 32 keys ----
    floatx8 s0 = {}, s1 = {};
    s0 = wmma_bf16(qa0, load_b_frag(lds_k, DK), s0);
    s0 = wmma_bf16(qa1, load_b_frag(lds_k + 32, DK), s0);
    s1 = wmma_bf16(qa0, load_b_frag(lds_k + 16 * DK, DK), s1);
    s1 = wmma_bf16(qa1, load_b_frag(lds_k + 16 * DK + 32, DK), s1);

    float rowm[8];
#pragma unroll
    for (int i = 0; i < 8; ++i) rowm[i] = fmaxf(s0[i], s1[i]);
#pragma unroll
    for (int i = 0; i < 8; ++i) {           // row max across 16 lanes per half
      float r = rowm[i];
      r = fmaxf(r, swz_xor<1>(r));
      r = fmaxf(r, swz_xor<2>(r));
      r = fmaxf(r, swz_xor<4>(r));
      r = fmaxf(r, swz_xor<8>(r));
      rowm[i] = r;
    }
    float al[8];
#pragma unroll
    for (int i = 0; i < 8; ++i) {
      float nm = fmaxf(m_i[i], rowm[i]);
      al[i] = exp2f(m_i[i] - nm);
      float e0 = exp2f(s0[i] - nm);
      float e1 = exp2f(s1[i] - nm);
      m_i[i] = nm;
#pragma unroll
      for (int n = 0; n < 4; ++n) acc[n][i] *= al[i];
      int row = i + ((lane >> 4) << 3);     // C layout -> row-major 16x32 in LDS
      lds_p[wave][row * 32 + (lane & 15)] = f2bf(e0);
      lds_p[wave][row * 32 + 16 + (lane & 15)] = f2bf(e1);
    }
    asm volatile("s_wait_dscnt 0" ::: "memory");   // intra-wave LDS RAW
    bf16x16 pa;
    {
      const bf16* pr = &lds_p[wave][(lane & 15) * 32 + ((lane >> 4) << 3)];
      pa = make_frag(*(const bf16x8*)pr, *(const bf16x8*)(pr + 16));
    }
    // ---- row sums of P via WMMA with all-ones B (replaces 32 shuffles) ----
    floatx8 rsum = {};
    rsum = wmma_bf16(pa, ones, rsum);
#pragma unroll
    for (int i = 0; i < 8; ++i) l_i[i] = l_i[i] * al[i] + rsum[i];

    // ---- acc += P @ V, B fragments from LDS V^T tile ----
#pragma unroll
    for (int n = 0; n < 4; ++n)
      acc[n] = wmma_bf16(pa, load_b_frag(lds_v + n * 16 * 32, 32), acc[n]);

    __syncthreads();   // protect K/V tiles before next iteration's overwrite
  }

  // ---- normalize, store attention output (bf16 [S][DMODEL], heads interleaved) --
#pragma unroll
  for (int i = 0; i < 8; ++i) {
    float inv = 1.0f / l_i[i];
#pragma unroll
    for (int n = 0; n < 4; ++n) acc[n][i] *= inv;
  }
  int rbase = qt * 16 + ((lane >> 4) << 3);
#pragma unroll
  for (int n = 0; n < 4; ++n) {
    int col = h * DK + n * 16 + (lane & 15);
#pragma unroll
    for (int i = 0; i < 8; ++i)
      O[(size_t)(rbase + i) * DMODEL + col] = f2bf(acc[n][i]);
  }
}

// ---------------- output projection: out = Oattn @ Wo^T + bo (fp32 out) ----------
__global__ void __launch_bounds__(256)
oproj_kernel(const bf16* __restrict__ X, const bf16* __restrict__ W,
             const float* __restrict__ bias, float* __restrict__ out) {
  int wave = threadIdx.x >> 5;
  int gid = blockIdx.x * 8 + wave;
  int tm = (gid >> 5) << 5;
  int tn = (gid & 31) << 5;
  const bf16* x0 = X + (size_t)tm * DMODEL;
  const bf16* x1 = x0 + 16 * DMODEL;
  const bf16* w0 = W + (size_t)tn * DMODEL;
  const bf16* w1 = w0 + 16 * DMODEL;

  floatx8 a00 = {}, a01 = {}, a10 = {}, a11 = {};
#pragma unroll 2
  for (int k = 0; k < DMODEL; k += 32) {
    bf16x16 fa0 = load_a_frag(x0 + k, DMODEL);
    bf16x16 fa1 = load_a_frag(x1 + k, DMODEL);
    bf16x16 fb0 = load_b_frag(w0 + k, DMODEL);
    bf16x16 fb1 = load_b_frag(w1 + k, DMODEL);
    a00 = wmma_bf16(fa0, fb0, a00);
    a01 = wmma_bf16(fa0, fb1, a01);
    a10 = wmma_bf16(fa1, fb0, a10);
    a11 = wmma_bf16(fa1, fb1, a11);
  }
  store_f32_tile(out, bias, a00, tm, tn);
  store_f32_tile(out, bias, a01, tm, tn + 16);
  store_f32_tile(out, bias, a10, tm + 16, tn);
  store_f32_tile(out, bias, a11, tm + 16, tn + 16);
}

extern "C" void kernel_launch(void* const* d_in, const int* in_sizes, int n_in,
                              void* d_out, int out_size, void* d_ws, size_t ws_size,
                              hipStream_t stream) {
  (void)in_sizes; (void)n_in; (void)out_size; (void)ws_size;
  const float* q_in = (const float*)d_in[0];
  const float* k_in = (const float*)d_in[1];
  const float* v_in = (const float*)d_in[2];
  const float* Wq   = (const float*)d_in[3];
  const float* bq   = (const float*)d_in[4];
  const float* Wk   = (const float*)d_in[5];
  const float* bk   = (const float*)d_in[6];
  const float* Wv   = (const float*)d_in[7];
  const float* bv   = (const float*)d_in[8];
  const float* Wo   = (const float*)d_in[9];
  const float* bo   = (const float*)d_in[10];
  float* out = (float*)d_out;

  const size_t SD = (size_t)S_LEN * DMODEL;   // 4M elems
  const size_t DD = (size_t)DMODEL * DMODEL;  // 1M elems
  char* ws = (char*)d_ws;
  bf16* xq  = (bf16*)ws;                 ws += SD * 2;
  bf16* xk  = (bf16*)ws;                 ws += SD * 2;
  bf16* xv  = (bf16*)ws;                 ws += SD * 2;
  bf16* wqb = (bf16*)ws;                 ws += DD * 2;
  bf16* wkb = (bf16*)ws;                 ws += DD * 2;
  bf16* wvb = (bf16*)ws;                 ws += DD * 2;
  bf16* wob = (bf16*)ws;                 ws += DD * 2;
  bf16* Qh  = (bf16*)ws;                 ws += SD * 2;   // [H][S][DK] (pre-scaled)
  bf16* Kh  = (bf16*)ws;                 ws += SD * 2;   // [H][S][DK]
  bf16* Vt  = (bf16*)ws;                 ws += SD * 2;   // [H][DK][S]
  bf16* Oat = (bf16*)ws;                 ws += SD * 2;   // [S][DMODEL]

  cast_bf16_kernel<<<(int)(SD / 8 / 256), 256, 0, stream>>>(q_in, xq, (int)(SD / 8));
  cast_bf16_kernel<<<(int)(SD / 8 / 256), 256, 0, stream>>>(k_in, xk, (int)(SD / 8));
  cast_bf16_kernel<<<(int)(SD / 8 / 256), 256, 0, stream>>>(v_in, xv, (int)(SD / 8));
  cast_bf16_kernel<<<(int)(DD / 8 / 256), 256, 0, stream>>>(Wq, wqb, (int)(DD / 8));
  cast_bf16_kernel<<<(int)(DD / 8 / 256), 256, 0, stream>>>(Wk, wkb, (int)(DD / 8));
  cast_bf16_kernel<<<(int)(DD / 8 / 256), 256, 0, stream>>>(Wv, wvb, (int)(DD / 8));
  cast_bf16_kernel<<<(int)(DD / 8 / 256), 256, 0, stream>>>(Wo, wob, (int)(DD / 8));

  // projections: (S/32)*(D/32) = 4096 wave-tiles, 8 waves/block -> 512 blocks
  const float qscale = 0.125f * 1.4426950408889634f;  // 1/sqrt(dk) * log2(e)
  proj_bf16_kernel<<<512, 256, 0, stream>>>(xq, wqb, bq, Qh, 0, qscale);
  proj_bf16_kernel<<<512, 256, 0, stream>>>(xk, wkb, bk, Kh, 0, 1.0f);
  proj_bf16_kernel<<<512, 256, 0, stream>>>(xv, wvb, bv, Vt, 1, 1.0f);

  // flash attention: 16 heads x 32 blocks (128 q rows each) = 512 blocks
  flash_attn_kernel<<<512, 256, 0, stream>>>(Qh, Kh, Vt, Oat);

  // output projection (fp32 out)
  oproj_kernel<<<512, 256, 0, stream>>>(Oat, wob, bo, out);
}
